// _SimpleDecoderLayer_88038239633781
// MI455X (gfx1250) — matrix operand order repaired and checked by reading, not compiled, not run on hardware
//
#include <hip/hip_runtime.h>

// ---------------- types ----------------
typedef __bf16 bf16_t;
typedef __attribute__((ext_vector_type(16))) __bf16 bf16x16;
typedef __attribute__((ext_vector_type(8)))  __bf16 bf16x8;
typedef __attribute__((ext_vector_type(8)))  float  f32x8;
typedef __attribute__((ext_vector_type(4)))  float  f32x4;

#define TOKENS 2048   // B*S
#define HDIM   1024
#define FDIM   2048
#define NEXP   8
#define SLOTS  4096   // TOKENS * topK

// ---------------- WMMA helpers ----------------
static __device__ __forceinline__ f32x8 wmma_bf16(bf16x16 a, bf16x16 b, f32x8 c) {
    // D = A(16x32) * B(32x16) + C, f32 accumulate
    return __builtin_amdgcn_wmma_f32_16x16x32_bf16(
        /*neg_a=*/false, a, /*neg_b=*/false, b,
        /*c_mod=*/(short)0, c, /*reuse_a=*/false, /*reuse_b=*/false);
}

// A fragment (16x32, bf16 source, row-major, K contiguous).
// lane r=lane%16 -> row r ; group g=lane/16 : elems[0..7]=K(k0+8g .. +7),
// elems[8..15]=K(k0+16+8g .. +7)   (per CDNA5 ISA A-layout table)
static __device__ __forceinline__ bf16x16 load_a_bf16(const bf16_t* __restrict__ row,
                                                      int k0, int g) {
    bf16x8 c0 = *(const bf16x8*)(row + k0 + 8 * g);
    bf16x8 c1 = *(const bf16x8*)(row + k0 + 16 + 8 * g);
    bf16x16 o;
#pragma unroll
    for (int i = 0; i < 8; ++i) { o[i] = c0[i]; o[8 + i] = c1[i]; }
    return o;
}

// A fragment from f32 source (convert to bf16 in registers)
static __device__ __forceinline__ bf16x16 load_a_f32(const float* __restrict__ row,
                                                     int k0, int g) {
    const float* p0 = row + k0 + 8 * g;
    const float* p1 = row + k0 + 16 + 8 * g;
    f32x4 a0 = *(const f32x4*)(p0);
    f32x4 a1 = *(const f32x4*)(p0 + 4);
    f32x4 b0 = *(const f32x4*)(p1);
    f32x4 b1 = *(const f32x4*)(p1 + 4);
    bf16x16 o;
#pragma unroll
    for (int i = 0; i < 4; ++i) {
        o[i]      = (__bf16)a0[i];
        o[4 + i]  = (__bf16)a1[i];
        o[8 + i]  = (__bf16)b0[i];
        o[12 + i] = (__bf16)b1[i];
    }
    return o;
}

// B fragment (32x16). Weight is [N,K] row-major -> column n of B = weight row n.
// lane r -> col n ; group g : elems[0..7]=K(k0+8g .. +7), elems[8..15]=K(k0+16+8g .. +7).
static __device__ __forceinline__ bf16x16 load_b_f32(const float* __restrict__ col,
                                                     int k0, int g) {
    const float* p = col + k0 + 8 * g;
    f32x4 q0 = *(const f32x4*)(p);
    f32x4 q1 = *(const f32x4*)(p + 4);
    f32x4 q2 = *(const f32x4*)(p + 16);
    f32x4 q3 = *(const f32x4*)(p + 20);
    bf16x16 o;
#pragma unroll
    for (int i = 0; i < 4; ++i) {
        o[i]      = (__bf16)q0[i];
        o[4 + i]  = (__bf16)q1[i];
        o[8 + i]  = (__bf16)q2[i];
        o[12 + i] = (__bf16)q3[i];
    }
    return o;
}

// ---------------- kernel 0: zero routing counters ----------------
__global__ void k_zero(int* counts) {
    if (threadIdx.x < NEXP) counts[threadIdx.x] = 0;
}

// ---------------- kernel 1: attn GEMM + residual -> h ----------------
// h[m,n] = hidden[m,n] + sum_k hidden[m,k]*attn_w[n,k] + attn_b[n] + 1e-4*noise[m,n]
// block tile 128x128, 4 waves (2x2), wave tile 64x64
__global__ __launch_bounds__(128) void k_attn(
    const float* __restrict__ hidden, const float* __restrict__ noise,
    const float* __restrict__ attn_w, const float* __restrict__ attn_b,
    float* __restrict__ h_f32, bf16_t* __restrict__ h_bf16) {
    const int lane = threadIdx.x & 31, wid = threadIdx.x >> 5;
    const int r = lane & 15, g = lane >> 4;
    const int m_w = blockIdx.y * 128 + (wid >> 1) * 64;
    const int n_w = blockIdx.x * 128 + (wid & 1) * 64;

    const float* arow[4];
    const float* bcol[4];
#pragma unroll
    for (int i = 0; i < 4; ++i) arow[i] = hidden + (size_t)(m_w + 16 * i + r) * HDIM;
#pragma unroll
    for (int j = 0; j < 4; ++j) bcol[j] = attn_w + (size_t)(n_w + 16 * j + r) * HDIM;

    f32x8 acc[4][4];
#pragma unroll
    for (int i = 0; i < 4; ++i)
#pragma unroll
        for (int j = 0; j < 4; ++j) acc[i][j] = {};

    for (int k0 = 0; k0 < HDIM; k0 += 32) {
        bf16x16 a[4], b[4];
#pragma unroll
        for (int i = 0; i < 4; ++i) a[i] = load_a_f32(arow[i], k0, g);
#pragma unroll
        for (int j = 0; j < 4; ++j) b[j] = load_b_f32(bcol[j], k0, g);
#pragma unroll
        for (int i = 0; i < 4; ++i)
#pragma unroll
            for (int j = 0; j < 4; ++j) acc[i][j] = wmma_bf16(a[i], b[j], acc[i][j]);
    }

#pragma unroll
    for (int i = 0; i < 4; ++i)
#pragma unroll
        for (int j = 0; j < 4; ++j) {
            const int n = n_w + 16 * j + r;
            const float bn = attn_b[n];
#pragma unroll
            for (int v = 0; v < 8; ++v) {
                const int m = m_w + 16 * i + v + 8 * g;  // C layout: M = vgpr + 8*(lane/16)
                const size_t idx = (size_t)m * HDIM + n;
                float hv = hidden[idx] + acc[i][j][v] + bn + 1e-4f * noise[idx];
                h_f32[idx] = hv;
                h_bf16[idx] = (bf16_t)hv;
            }
        }
}

// ---------------- kernel 2: router (one wave per token) ----------------
__global__ __launch_bounds__(128) void k_router(
    const float* __restrict__ h, const float* __restrict__ gate_w,
    const float* __restrict__ gate_b, int* __restrict__ counts,
    int* __restrict__ tok_of, int* __restrict__ top_e,
    int* __restrict__ top_pos, float* __restrict__ top_w) {
    const int lane = threadIdx.x & 31, wid = threadIdx.x >> 5;
    const int t = blockIdx.x * 4 + wid;
    const float* hp = h + (size_t)t * HDIM;

    float acc[NEXP];
#pragma unroll
    for (int e = 0; e < NEXP; ++e) acc[e] = 0.f;
    for (int i = lane; i < HDIM; i += 32) {
        float x = hp[i];
#pragma unroll
        for (int e = 0; e < NEXP; ++e) acc[e] += x * gate_w[e * HDIM + i];
    }
#pragma unroll
    for (int e = 0; e < NEXP; ++e)
        for (int off = 16; off > 0; off >>= 1) acc[e] += __shfl_xor(acc[e], off, 32);

    if (lane == 0) {
        float lg[NEXP];
#pragma unroll
        for (int e = 0; e < NEXP; ++e) lg[e] = acc[e] + gate_b[e];
        int i0 = 0;
#pragma unroll
        for (int e = 1; e < NEXP; ++e) if (lg[e] > lg[i0]) i0 = e;
        int i1 = (i0 == 0) ? 1 : 0;
#pragma unroll
        for (int e = 0; e < NEXP; ++e) if (e != i0 && lg[e] > lg[i1]) i1 = e;
        // top-2 softmax renormalized == 2-way softmax over selected logits
        float w0 = 1.f / (1.f + __expf(lg[i1] - lg[i0]));
        float w1 = 1.f - w0;
        int p0 = atomicAdd(&counts[i0], 1);
        int p1 = atomicAdd(&counts[i1], 1);
        tok_of[i0 * TOKENS + p0] = t;
        tok_of[i1 * TOKENS + p1] = t;
        top_e[2 * t] = i0;   top_e[2 * t + 1] = i1;
        top_pos[2 * t] = p0; top_pos[2 * t + 1] = p1;
        top_w[2 * t] = w0;   top_w[2 * t + 1] = w1;
    }
}

// ---------------- kernel 3: exclusive prefix over 8 counts ----------------
__global__ void k_offsets(const int* __restrict__ counts, int* __restrict__ offsets) {
    if (threadIdx.x == 0) {
        int o = 0;
        for (int e = 0; e < NEXP; ++e) { offsets[e] = o; o += counts[e]; }
    }
}

// ---------------- kernel 4: fused w1/w3 GEMM + SwiGLU -> act (bf16) ---------
// grid: x = F/128, y = Mtiles(64), z = expert. block 64x128 (4 waves, wave 32x64)
__global__ __launch_bounds__(128) void k_expert_up(
    const bf16_t* __restrict__ hb, const float* __restrict__ w1,
    const float* __restrict__ w3, const int* __restrict__ counts,
    const int* __restrict__ offsets, const int* __restrict__ tok_of,
    bf16_t* __restrict__ act) {
    const int e = blockIdx.z;
    const int Te = counts[e];
    const int m_blk = blockIdx.y * 64;
    if (m_blk >= Te) return;
    const int lane = threadIdx.x & 31, wid = threadIdx.x >> 5;
    const int r = lane & 15, g = lane >> 4;
    const int m_w = m_blk + (wid >> 1) * 32;
    const int n_w = blockIdx.x * 128 + (wid & 1) * 64;
    const int offE = offsets[e];
    const float* w1e = w1 + (size_t)e * FDIM * HDIM;
    const float* w3e = w3 + (size_t)e * FDIM * HDIM;

    const bf16_t* arow[2];
#pragma unroll
    for (int i = 0; i < 2; ++i) {
        int rw = m_w + 16 * i + r;
        int rc = rw < Te ? rw : Te - 1;  // clamp gather for partial tiles
        arow[i] = hb + (size_t)tok_of[e * TOKENS + rc] * HDIM;
    }
    const float* b1col[4];
    const float* b3col[4];
#pragma unroll
    for (int j = 0; j < 4; ++j) {
        b1col[j] = w1e + (size_t)(n_w + 16 * j + r) * HDIM;
        b3col[j] = w3e + (size_t)(n_w + 16 * j + r) * HDIM;
    }

    f32x8 accg[2][4], accu[2][4];
#pragma unroll
    for (int i = 0; i < 2; ++i)
#pragma unroll
        for (int j = 0; j < 4; ++j) { accg[i][j] = {}; accu[i][j] = {}; }

    for (int k0 = 0; k0 < HDIM; k0 += 32) {
        bf16x16 a[2];
#pragma unroll
        for (int i = 0; i < 2; ++i) a[i] = load_a_bf16(arow[i], k0, g);
#pragma unroll
        for (int j = 0; j < 4; ++j) {
            bf16x16 b1f = load_b_f32(b1col[j], k0, g);
#pragma unroll
            for (int i = 0; i < 2; ++i) accg[i][j] = wmma_bf16(a[i], b1f, accg[i][j]);
            bf16x16 b3f = load_b_f32(b3col[j], k0, g);
#pragma unroll
            for (int i = 0; i < 2; ++i) accu[i][j] = wmma_bf16(a[i], b3f, accu[i][j]);
        }
    }

#pragma unroll
    for (int i = 0; i < 2; ++i)
#pragma unroll
        for (int j = 0; j < 4; ++j) {
            const int f = n_w + 16 * j + r;
#pragma unroll
            for (int v = 0; v < 8; ++v) {
                const int m = m_w + 16 * i + v + 8 * g;
                if (m < Te) {
                    float gv = accg[i][j][v], uv = accu[i][j][v];
                    float s = gv / (1.f + __expf(-gv)) * uv;  // silu(g)*u
                    act[(size_t)(offE + m) * FDIM + f] = (bf16_t)s;
                }
            }
        }
}

// ---------------- kernel 5: w2 GEMM -> y (f32, per-slot) ----------------
// grid: x = H/128, y = Mtiles(128), z = expert. block 128x128, wave 64x64
__global__ __launch_bounds__(128) void k_expert_down(
    const bf16_t* __restrict__ act, const float* __restrict__ w2,
    const int* __restrict__ counts, const int* __restrict__ offsets,
    float* __restrict__ y) {
    const int e = blockIdx.z;
    const int Te = counts[e];
    const int m_blk = blockIdx.y * 128;
    if (m_blk >= Te) return;
    const int lane = threadIdx.x & 31, wid = threadIdx.x >> 5;
    const int r = lane & 15, g = lane >> 4;
    const int m_w = m_blk + (wid >> 1) * 64;
    const int n_w = blockIdx.x * 128 + (wid & 1) * 64;
    const int offE = offsets[e];
    const bf16_t* Ae = act + (size_t)offE * FDIM;  // compact rows, no gather
    const float* w2e = w2 + (size_t)e * HDIM * FDIM;

    const bf16_t* arow[4];
    const float* bcol[4];
#pragma unroll
    for (int i = 0; i < 4; ++i) {
        int rw = m_w + 16 * i + r;
        int rc = rw < Te ? rw : Te - 1;
        arow[i] = Ae + (size_t)rc * FDIM;
    }
#pragma unroll
    for (int j = 0; j < 4; ++j) bcol[j] = w2e + (size_t)(n_w + 16 * j + r) * FDIM;

    f32x8 acc[4][4];
#pragma unroll
    for (int i = 0; i < 4; ++i)
#pragma unroll
        for (int j = 0; j < 4; ++j) acc[i][j] = {};

    for (int k0 = 0; k0 < FDIM; k0 += 32) {
        bf16x16 a[4], b[4];
#pragma unroll
        for (int i = 0; i < 4; ++i) a[i] = load_a_bf16(arow[i], k0, g);
#pragma unroll
        for (int j = 0; j < 4; ++j) b[j] = load_b_f32(bcol[j], k0, g);
#pragma unroll
        for (int i = 0; i < 4; ++i)
#pragma unroll
            for (int j = 0; j < 4; ++j) acc[i][j] = wmma_bf16(a[i], b[j], acc[i][j]);
    }

#pragma unroll
    for (int i = 0; i < 4; ++i)
#pragma unroll
        for (int j = 0; j < 4; ++j) {
            const int n = n_w + 16 * j + r;
#pragma unroll
            for (int v = 0; v < 8; ++v) {
                const int m = m_w + 16 * i + v + 8 * g;
                if (m < Te) y[(size_t)(offE + m) * HDIM + n] = acc[i][j][v];
            }
        }
}

// ---------------- kernel 6: deterministic combine ----------------
__global__ __launch_bounds__(256) void k_combine(
    const float* __restrict__ hf, const float* __restrict__ y,
    const int* __restrict__ offsets, const int* __restrict__ top_e,
    const int* __restrict__ top_pos, const float* __restrict__ top_w,
    float* __restrict__ out) {
    const int t = blockIdx.x;
    const int r0 = offsets[top_e[2 * t]] + top_pos[2 * t];
    const int r1 = offsets[top_e[2 * t + 1]] + top_pos[2 * t + 1];
    const float w0 = top_w[2 * t], w1 = top_w[2 * t + 1];
    const float* y0 = y + (size_t)r0 * HDIM;
    const float* y1 = y + (size_t)r1 * HDIM;
    const float* hp = hf + (size_t)t * HDIM;
    float* op = out + (size_t)t * HDIM;
    for (int n = threadIdx.x; n < HDIM; n += 256)
        op[n] = hp[n] + w0 * y0[n] + w1 * y1[n];
}

// ---------------- launch ----------------
extern "C" void kernel_launch(void* const* d_in, const int* in_sizes, int n_in,
                              void* d_out, int out_size, void* d_ws, size_t ws_size,
                              hipStream_t stream) {
    const float* hidden = (const float*)d_in[0];
    const float* noise  = (const float*)d_in[1];
    const float* attn_w = (const float*)d_in[2];
    const float* attn_b = (const float*)d_in[3];
    const float* gate_w = (const float*)d_in[4];
    const float* gate_b = (const float*)d_in[5];
    const float* w1     = (const float*)d_in[6];
    const float* w3     = (const float*)d_in[7];
    const float* w2     = (const float*)d_in[8];
    float* out = (float*)d_out;

    char* w = (char*)d_ws;
    float*  h_f32  = (float*)w;  w += (size_t)TOKENS * HDIM * sizeof(float);   // 8 MB
    bf16_t* h_bf16 = (bf16_t*)w; w += (size_t)TOKENS * HDIM * sizeof(bf16_t);  // 4 MB
    bf16_t* act    = (bf16_t*)w; w += (size_t)SLOTS * FDIM * sizeof(bf16_t);   // 16 MB
    float*  y      = (float*)w;  w += (size_t)SLOTS * HDIM * sizeof(float);    // 16 MB
    int* counts  = (int*)w; w += 256;
    int* offsets = (int*)w; w += 256;
    int* tok_of  = (int*)w; w += (size_t)NEXP * TOKENS * sizeof(int);
    int* top_e   = (int*)w; w += (size_t)TOKENS * 2 * sizeof(int);
    int* top_pos = (int*)w; w += (size_t)TOKENS * 2 * sizeof(int);
    float* top_w = (float*)w; w += (size_t)TOKENS * 2 * sizeof(float);

    k_zero<<<1, 32, 0, stream>>>(counts);
    k_attn<<<dim3(HDIM / 128, TOKENS / 128), 128, 0, stream>>>(
        hidden, noise, attn_w, attn_b, h_f32, h_bf16);
    k_router<<<TOKENS / 4, 128, 0, stream>>>(
        h_f32, gate_w, gate_b, counts, tok_of, top_e, top_pos, top_w);
    k_offsets<<<1, 32, 0, stream>>>(counts, offsets);
    k_expert_up<<<dim3(FDIM / 128, TOKENS / 64, NEXP), 128, 0, stream>>>(
        h_bf16, w1, w3, counts, offsets, tok_of, act);
    k_expert_down<<<dim3(HDIM / 128, TOKENS / 128, NEXP), 128, 0, stream>>>(
        act, w2, counts, offsets, y);
    k_combine<<<TOKENS, 256, 0, stream>>>(
        h_f32, y, offsets, top_e, top_pos, top_w, out);
}
